// BlockTridiagonalWhitening_38860864094620
// MI455X (gfx1250) — matrix operand order repaired and checked by reading, not compiled, not run on hardware
//
#include <hip/hip_runtime.h>
#include <hip/hip_bf16.h>
#include <stdint.h>

// ---------------------------------------------------------------------------
// BlockTridiagonalWhitening  (B=64, T=2048, C=256)
//   w_t = (x_t - mean(x_t)) @ V0  -  w_{t-1} @ (V1 @ V0)
// Rewritten as w_t = y_t + w_{t-1} @ N,  N = -(V1 @ V0):
//   K_prep : N^(2^j) f32 powers + bf16 WMMA-B-layout packs of V0/N/N2/N4/N8/N16
//   K_y    : y_t = center(x_t) @ V0                 (parallel WMMA GEMM)
//   K_pass : y'_t = y_t + y_{t-d} @ N^d, d=1,2,4,8  (parallel, ping-pong)
//   K_scan : w_t = y''''_t + w_{t-16} @ N^16        (64 chains of depth 128)
// All matmuls: v_wmma_f32_16x16x32_bf16, M=16 batches, N=16 ch, K=256.
// Roofline: passes are bandwidth-bound sweeps (~12us each @ 23.3 TB/s);
// the serial phase is latency-bound (~200ns/step), so 4 levels (k=4)
// minimizes k*Pass + (T/2^k)*StepLatency.
// ---------------------------------------------------------------------------

typedef __attribute__((ext_vector_type(16))) __bf16 v16bf;
typedef __attribute__((ext_vector_type(8)))  __bf16 v8bf;
typedef __attribute__((ext_vector_type(8)))  float  v8f;

#define DEV __device__ __forceinline__

constexpr int Bn  = 64;
constexpr int Tn  = 2048;
constexpr int Cn  = 256;
constexpr int BGn = Bn / 16;     // 4 batch groups (M tiles)
constexpr int KFn = Cn / 32;     // 8 K-fragments of 32
constexpr int STRIDE = 16;       // scan stride after 4 fold levels
constexpr int PITCH = Cn + 8;    // LDS row pitch (bf16) to break bank conflicts

// Native converts: let clang emit v_cvt_pk_bf16_f32 / shift, not a SW RNE seq.
DEV __bf16 f2bf(float f) { return (__bf16)f; }
DEV float  bf2f(__bf16 b) { return (float)b; }
DEV uint16_t f2bfbits(float f) {
  union { __bf16 b; uint16_t s; } y; y.b = (__bf16)f; return y.s;
}

// --------------------------------------------------------------------------
// Prep: N = -(V1@V0); N2..N16 by repeated squaring (f32, one WG), then pack
// each B-operand into the wave32 WMMA bf16 B-lane layout:
//   pack[((kt*16+nt)*32+lane)*16 + h] = M[kt*32 + 8*(lane>=16) + (h&7) + 16*(h>=8)][nt*16 + lane%16]
// so kernels fetch one 32B contiguous v16bf per lane per fragment.
// --------------------------------------------------------------------------
DEV void gemm_slice(const float* A, const float* Bm, float* D, float sgn, int tid, int nthr) {
  for (int o = tid; o < Cn * Cn; o += nthr) {
    int r = o >> 8, c = o & 255;
    float acc = 0.f;
    for (int k = 0; k < Cn; ++k) acc = fmaf(A[r * Cn + k], Bm[k * Cn + c], acc);
    D[o] = sgn * acc;
  }
}
DEV void pack_b(const float* src, uint16_t* dst, int tid, int nthr) {
  for (int e = tid; e < Cn * Cn; e += nthr) {
    int h = e & 15, lane = (e >> 4) & 31, tile = e >> 9;
    int kt = tile >> 4, nt = tile & 15;
    int k = kt * 32 + ((lane >> 4) << 3) + (h & 7) + ((h >> 3) << 4);
    int n = (nt << 4) + (lane & 15);
    dst[e] = f2bfbits(src[k * Cn + n]);
  }
}

__global__ __launch_bounds__(1024) void k_prep(const float* __restrict__ V0,
                                               const float* __restrict__ V1,
                                               float* nf1, float* nf2, float* nf4,
                                               float* nf8, float* nf16,
                                               uint16_t* pkV0, uint16_t* pkN1,
                                               uint16_t* pkN2, uint16_t* pkN4,
                                               uint16_t* pkN8, uint16_t* pkN16) {
  const int tid = threadIdx.x, nthr = blockDim.x;
  gemm_slice(V1, V0, nf1, -1.f, tid, nthr);          // N = -(V1@V0)
  __threadfence(); __syncthreads();
  gemm_slice(nf1, nf1, nf2, 1.f, tid, nthr);         // N^2
  __threadfence(); __syncthreads();
  gemm_slice(nf2, nf2, nf4, 1.f, tid, nthr);         // N^4
  __threadfence(); __syncthreads();
  gemm_slice(nf4, nf4, nf8, 1.f, tid, nthr);         // N^8
  __threadfence(); __syncthreads();
  gemm_slice(nf8, nf8, nf16, 1.f, tid, nthr);        // N^16
  __threadfence(); __syncthreads();
  pack_b(V0, pkV0, tid, nthr);
  pack_b(nf1, pkN1, tid, nthr);
  pack_b(nf2, pkN2, tid, nthr);
  pack_b(nf4, pkN4, tid, nthr);
  pack_b(nf8, pkN8, tid, nthr);
  pack_b(nf16, pkN16, tid, nthr);
}

// Load the 8 resident B fragments for this wave's N-tile.
DEV void load_bfrags(const uint16_t* pk, int nt, int lane, v16bf bf[KFn]) {
#pragma unroll
  for (int f = 0; f < KFn; ++f)
    bf[f] = *reinterpret_cast<const v16bf*>(pk + (((f * 16 + nt) * 32 + lane) << 4));
}

// Build a bf16 A fragment (16x32) from two 16B chunks of a row-major bf16 row.
DEV v16bf afrag_bf16(const __bf16* arow, int f, int lh) {
  const int c0 = f * 32 + lh * 8;
  v8bf lo = *reinterpret_cast<const v8bf*>(arow + c0);
  v8bf hi = *reinterpret_cast<const v8bf*>(arow + c0 + 16);
  v16bf a;
#pragma unroll
  for (int i = 0; i < 8; ++i) { a[i] = lo[i]; a[i + 8] = hi[i]; }
  return a;
}

// --------------------------------------------------------------------------
// K_y: y[t,b,:] = (x[b,t,:] - mean) @ V0, written bf16 to bufA[t][b][c].
// Grid (T, BG), 512 thr; wave = N-tile. Mean fused via intra-row shfl_xor(16).
// --------------------------------------------------------------------------
__global__ __launch_bounds__(512) void k_y(const float* __restrict__ x,
                                           const uint16_t* __restrict__ pkV0,
                                           __bf16* __restrict__ dst) {
  const int t = blockIdx.x, bg = blockIdx.y;
  const int lane = threadIdx.x & 31, nt = threadIdx.x >> 5;
  const int lh = lane >> 4, lm = lane & 15;

  v16bf bf[KFn];
  load_bfrags(pkV0, nt, lane, bf);

  const float* arow = x + ((size_t)(bg * 16 + lm) * Tn + t) * Cn;

  float s = 0.f;                                    // pass 1: row-sum
#pragma unroll
  for (int f = 0; f < KFn; ++f) {
    const int c0 = f * 32 + lh * 8;
    float4 q0 = *reinterpret_cast<const float4*>(arow + c0);
    float4 q1 = *reinterpret_cast<const float4*>(arow + c0 + 4);
    float4 q2 = *reinterpret_cast<const float4*>(arow + c0 + 16);
    float4 q3 = *reinterpret_cast<const float4*>(arow + c0 + 20);
    s += q0.x + q0.y + q0.z + q0.w + q1.x + q1.y + q1.z + q1.w +
         q2.x + q2.y + q2.z + q2.w + q3.x + q3.y + q3.z + q3.w;
  }
  const float mean = (s + __shfl_xor(s, 16, 32)) * (1.f / Cn);

  v8f acc = 0.0f;                                   // pass 2: center+cvt+WMMA
#pragma unroll
  for (int f = 0; f < KFn; ++f) {
    const int c0 = f * 32 + lh * 8;
    float4 q0 = *reinterpret_cast<const float4*>(arow + c0);
    float4 q1 = *reinterpret_cast<const float4*>(arow + c0 + 4);
    float4 q2 = *reinterpret_cast<const float4*>(arow + c0 + 16);
    float4 q3 = *reinterpret_cast<const float4*>(arow + c0 + 20);
    float va[16] = {q0.x, q0.y, q0.z, q0.w, q1.x, q1.y, q1.z, q1.w,
                    q2.x, q2.y, q2.z, q2.w, q3.x, q3.y, q3.z, q3.w};
    v16bf a;
#pragma unroll
    for (int i = 0; i < 16; ++i) a[i] = f2bf(va[i] - mean);
    acc = __builtin_amdgcn_wmma_f32_16x16x32_bf16(false, a, false, bf[f],
                                                  (short)0, acc, false, false);
  }
  __bf16* drow = dst + ((size_t)t * Bn + bg * 16) * Cn + nt * 16 + lm;
#pragma unroll
  for (int v = 0; v < 8; ++v) drow[(size_t)(v + 8 * lh) * Cn] = f2bf(acc[v]);
}

// --------------------------------------------------------------------------
// K_pass: dst[t] = src[t] + src[t-delta] @ N^delta (t >= delta, else copy).
// --------------------------------------------------------------------------
__global__ __launch_bounds__(512) void k_pass(const __bf16* __restrict__ src,
                                              __bf16* __restrict__ dst,
                                              const uint16_t* __restrict__ pk,
                                              int delta) {
  const int t = blockIdx.x, bg = blockIdx.y;
  const size_t tile = ((size_t)t * Bn + bg * 16) * Cn;
  if (t < delta) {                                   // verbatim copy (8KB tile)
    const uint4* s = reinterpret_cast<const uint4*>(src + tile);
    uint4* d = reinterpret_cast<uint4*>(dst + tile);
    d[threadIdx.x] = s[threadIdx.x];
    return;
  }
  const int lane = threadIdx.x & 31, nt = threadIdx.x >> 5;
  const int lh = lane >> 4, lm = lane & 15;

  v16bf bf[KFn];
  load_bfrags(pk, nt, lane, bf);

  const __bf16* arow = src + ((size_t)(t - delta) * Bn + bg * 16 + lm) * Cn;
  const __bf16* crow = src + tile + nt * 16 + lm;

  v8f acc;
#pragma unroll
  for (int v = 0; v < 8; ++v) acc[v] = bf2f(crow[(size_t)(v + 8 * lh) * Cn]);
#pragma unroll
  for (int f = 0; f < KFn; ++f)
    acc = __builtin_amdgcn_wmma_f32_16x16x32_bf16(false, afrag_bf16(arow, f, lh),
                                                  false, bf[f], (short)0, acc,
                                                  false, false);
  __bf16* drow = dst + tile + nt * 16 + lm;
#pragma unroll
  for (int v = 0; v < 8; ++v) drow[(size_t)(v + 8 * lh) * Cn] = f2bf(acc[v]);
}

// --------------------------------------------------------------------------
// K_scan: w_t = z_t + w_{t-16} @ N^16.  64 WGs = 4 batch groups x 16 residues,
// 16 waves (one per N-tile). State double-buffered in LDS (bf16), A-frags via
// ds_load_b128, resident N^16 B-frags, one s_barrier per step, z software-
// pipelined + prefetched. Writes f32 result to d_out (B,T,C).
// --------------------------------------------------------------------------
__global__ __launch_bounds__(512) void k_scan(const __bf16* __restrict__ z,
                                              const uint16_t* __restrict__ pkNS,
                                              float* __restrict__ out) {
  const int bg = blockIdx.x >> 4, r = blockIdx.x & 15;
  const int lane = threadIdx.x & 31, nt = threadIdx.x >> 5;
  const int lh = lane >> 4, lm = lane & 15;

  __shared__ __align__(16) __bf16 wlds[2][16][PITCH];
  {
    uint16_t* p = reinterpret_cast<uint16_t*>(&wlds[0][0][0]);
    for (int i = threadIdx.x; i < 2 * 16 * PITCH; i += 512) p[i] = 0;
  }

  v16bf bf[KFn];
  load_bfrags(pkNS, nt, lane, bf);
  __syncthreads();

  // preload z tile for t = r
  float zc[8];
  {
    const __bf16* zrow = z + ((size_t)r * Bn + bg * 16) * Cn + nt * 16 + lm;
#pragma unroll
    for (int v = 0; v < 8; ++v) zc[v] = bf2f(zrow[(size_t)(v + 8 * lh) * Cn]);
  }

  int cur = 0;
  for (int it = 0; it < Tn / STRIDE; ++it) {
    const int t = r + STRIDE * it;
    v8f acc;
#pragma unroll
    for (int v = 0; v < 8; ++v) acc[v] = zc[v];

#pragma unroll
    for (int f = 0; f < KFn; ++f) {                 // A from LDS state
      const int c0 = f * 32 + lh * 8;
      v8bf lo = *reinterpret_cast<const v8bf*>(&wlds[cur][lm][c0]);
      v8bf hi = *reinterpret_cast<const v8bf*>(&wlds[cur][lm][c0 + 16]);
      v16bf a;
#pragma unroll
      for (int i = 0; i < 8; ++i) { a[i] = lo[i]; a[i + 8] = hi[i]; }
      acc = __builtin_amdgcn_wmma_f32_16x16x32_bf16(false, a, false, bf[f],
                                                    (short)0, acc, false, false);
    }

    // emit f32 result + publish bf16 state for the next step
#pragma unroll
    for (int v = 0; v < 8; ++v) {
      out[((size_t)(bg * 16 + v + 8 * lh) * Tn + t) * Cn + nt * 16 + lm] = acc[v];
      wlds[cur ^ 1][v + 8 * lh][nt * 16 + lm] = f2bf(acc[v]);
    }

    if (t + STRIDE < Tn) {                          // pipeline next z tile
      const __bf16* zn =
          z + ((size_t)(t + STRIDE) * Bn + bg * 16) * Cn + nt * 16 + lm;
#pragma unroll
      for (int v = 0; v < 8; ++v) zc[v] = bf2f(zn[(size_t)(v + 8 * lh) * Cn]);
      if (t + 2 * STRIDE < Tn)
        __builtin_prefetch(z + ((size_t)(t + 2 * STRIDE) * Bn + bg * 16) * Cn, 0, 1);
    }
    __syncthreads();
    cur ^= 1;
  }
}

// --------------------------------------------------------------------------
extern "C" void kernel_launch(void* const* d_in, const int* in_sizes, int n_in,
                              void* d_out, int out_size, void* d_ws, size_t ws_size,
                              hipStream_t stream) {
  (void)in_sizes; (void)n_in; (void)out_size; (void)ws_size;
  const float* x  = (const float*)d_in[0];
  const float* V0 = (const float*)d_in[1];
  const float* V1 = (const float*)d_in[2];
  float* out = (float*)d_out;
  char*  ws  = (char*)d_ws;

  // workspace layout (~132 MB): 6x 128KB bf16 B-packs, 5x 256KB f32 powers,
  // 2x 64MB bf16 ping-pong buffers.
  uint16_t* pkV0  = (uint16_t*)(ws + (0 << 10));
  uint16_t* pkN1  = (uint16_t*)(ws + (128 << 10));
  uint16_t* pkN2  = (uint16_t*)(ws + (256 << 10));
  uint16_t* pkN4  = (uint16_t*)(ws + (384 << 10));
  uint16_t* pkN8  = (uint16_t*)(ws + (512 << 10));
  uint16_t* pkN16 = (uint16_t*)(ws + (640 << 10));
  float*    nf1   = (float*)(ws + (1024 << 10));
  float*    nf2   = (float*)(ws + (1280 << 10));
  float*    nf4   = (float*)(ws + (1536 << 10));
  float*    nf8   = (float*)(ws + (1792 << 10));
  float*    nf16  = (float*)(ws + (2048 << 10));
  const size_t bufBytes = (size_t)Tn * Bn * Cn * sizeof(uint16_t);   // 64 MB
  __bf16* bufA = (__bf16*)(ws + (4ull << 20));
  __bf16* bufB = (__bf16*)(ws + (4ull << 20) + bufBytes);

  k_prep<<<1, 1024, 0, stream>>>(V0, V1, nf1, nf2, nf4, nf8, nf16,
                                 pkV0, pkN1, pkN2, pkN4, pkN8, pkN16);

  dim3 grid(Tn, BGn);
  k_y   <<<grid, 512, 0, stream>>>(x, pkV0, bufA);        // y = center(x)@V0
  k_pass<<<grid, 512, 0, stream>>>(bufA, bufB, pkN1, 1);  // fold t-1  (N)
  k_pass<<<grid, 512, 0, stream>>>(bufB, bufA, pkN2, 2);  // fold t-2  (N^2)
  k_pass<<<grid, 512, 0, stream>>>(bufA, bufB, pkN4, 4);  // fold t-4  (N^4)
  k_pass<<<grid, 512, 0, stream>>>(bufB, bufA, pkN8, 8);  // fold t-8  (N^8)
  k_scan<<<64,   512, 0, stream>>>(bufA, pkN16, out);     // stride-16 scan
}
